// UFLAttention_19533511262510
// MI455X (gfx1250) — compile-verified, matrix operand order checked by reading, not run on hardware
//
#include <hip/hip_runtime.h>
#include <hip/hip_fp16.h>

typedef __attribute__((ext_vector_type(2)))  float    v2f;
typedef __attribute__((ext_vector_type(8)))  float    v8f;
typedef __attribute__((ext_vector_type(16))) _Float16 v16h;

#define B_SZ 512
#define D_SZ 256
#define H_SZ 8
#define YS   260   // LDS row stride in floats: 1040B -> 16B aligned rows, bank-spread

// One wave (32 threads) per block, 16 batch rows per block, grid = 32.
// y[b,i] = relu(x[b,i]*(1 + sum_h Wv[i,h]/16) + sum_h bv[i,h]/16)  (exact softmax collapse)
// out    = layernorm_D(y) * gamma + beta
// Layernorm stats (sum, sumsq over D=256) computed on the matrix pipe:
// C += A(16x4 f32 tile of y) x B(ones)  via V_WMMA_F32_16X16X4_F32.
__global__ __launch_bounds__(32) void ufl_attn_ln_kernel(
    const float* __restrict__ x,
    const float* __restrict__ Wv,
    const float* __restrict__ bv,
    const float* __restrict__ gamma,
    const float* __restrict__ beta,
    float* __restrict__ out)
{
    __shared__ float s_scale[D_SZ];
    __shared__ float s_shift[D_SZ];
    __shared__ float s_y[16 * YS];
    __shared__ float s_sum[16];
    __shared__ float s_sq[16];

    const int tid     = threadIdx.x;        // 0..31 (one wave32)
    const int rowbase = blockIdx.x * 16;

    // gfx1250 prefetch path (lowers to global_prefetch_b8)
    __builtin_prefetch(x + (size_t)rowbase * D_SZ, 0, 0);

    // ---- per-feature scale/shift from head-sums of Wv / bv ----
#pragma unroll
    for (int j = 0; j < 8; ++j) {
        const int i = tid + 32 * j;
        const float4* wv4 = (const float4*)(Wv + i * H_SZ);
        const float4* bv4 = (const float4*)(bv + i * H_SZ);
        float4 w0 = wv4[0], w1 = wv4[1];
        float4 b0 = bv4[0], b1 = bv4[1];
        float sw = (w0.x + w0.y) + (w0.z + w0.w) + (w1.x + w1.y) + (w1.z + w1.w);
        float sb = (b0.x + b0.y) + (b0.z + b0.w) + (b1.x + b1.y) + (b1.z + b1.w);
        s_scale[i] = 1.0f + 0.0625f * sw;   // 1/sqrt(D) = 1/16 exactly
        s_shift[i] = 0.0625f * sb;
    }
    __syncthreads();

    // ---- stage y = relu(x*scale + shift) tile into LDS (coalesced b128 loads) ----
#pragma unroll 4
    for (int j = 0; j < 32; ++j) {
        const int idx = j * 32 + tid;       // float4 index within 16x256 tile
        const int r   = idx >> 6;           // 64 float4 per row
        const int c   = (idx & 63) * 4;
        float4 xv = *(const float4*)(x + (size_t)(rowbase + r) * D_SZ + c);
        float4 sc = *(const float4*)&s_scale[c];
        float4 sh = *(const float4*)&s_shift[c];
        float* yd = &s_y[r * YS + c];
        yd[0] = fmaxf(0.0f, xv.x * sc.x + sh.x);
        yd[1] = fmaxf(0.0f, xv.y * sc.y + sh.y);
        yd[2] = fmaxf(0.0f, xv.z * sc.z + sh.z);
        yd[3] = fmaxf(0.0f, xv.w * sc.w + sh.w);
    }
    __syncthreads();

    // ---- row reductions (sum, sumsq) on the matrix pipe ----
    const int l    = tid & 15;              // A-matrix row owned by this lane
    const int half = tid >> 4;              // lane-half selects K pair
    const float* yrow = &s_y[l * YS];

    v8f csum = {};
    v8f csq  = {};

#if __has_builtin(__builtin_amdgcn_wmma_f32_16x16x4_f32)
    // Exact f32 path: A = 16x4 chunk of y (2 VGPRs: lane<16 holds K=0,1; lane>=16 K=2,3),
    // B = ones (layout-irrelevant), C accumulates row sums replicated across N.
    const v2f bones = {1.0f, 1.0f};
    for (int k0 = 0; k0 < D_SZ; k0 += 4) {
        float y0 = yrow[k0 + 2 * half];
        float y1 = yrow[k0 + 2 * half + 1];
        v2f a  = {y0, y1};
        v2f a2 = {y0 * y0, y1 * y1};
        csum = __builtin_amdgcn_wmma_f32_16x16x4_f32(false, a,  false, bones,
                                                     (short)0, csum, false, false);
        csq  = __builtin_amdgcn_wmma_f32_16x16x4_f32(false, a2, false, bones,
                                                     (short)0, csq,  false, false);
    }
#else
    // Fallback (probe-confirmed builtin): f16 A, f32 accumulate, 16x16x32.
    v16h bones;
#pragma unroll
    for (int i = 0; i < 16; ++i) bones[i] = (_Float16)1.0f;
    for (int k0 = 0; k0 < D_SZ; k0 += 32) {
        v16h a, a2;
#pragma unroll
        for (int j = 0; j < 4; ++j) {
#pragma unroll
            for (int c = 0; c < 2; ++c) {
                float lo = yrow[k0 + 8 * half + 2 * j + c];
                float hi = yrow[k0 + 16 + 8 * half + 2 * j + c];
                a [2 * j + c]     = (_Float16)lo;
                a [8 + 2 * j + c] = (_Float16)hi;
                a2[2 * j + c]     = (_Float16)(lo * lo);
                a2[8 + 2 * j + c] = (_Float16)(hi * hi);
            }
        }
        csum = __builtin_amdgcn_wmma_f32_16x16x32_f16(false, a,  false, bones,
                                                      (short)0, csum, false, false);
        csq  = __builtin_amdgcn_wmma_f32_16x16x32_f16(false, a2, false, bones,
                                                      (short)0, csq,  false, false);
    }
#endif

    // C layout: VGPR j, lanes 0-15 -> (M=j, N=lane); lanes 16-31 -> (M=j+8, N=lane-16).
    // N=0 lives on lane 0 (rows 0..7) and lane 16 (rows 8..15).
    if ((tid & 15) == 0) {
        const int base = half * 8;
#pragma unroll
        for (int j = 0; j < 8; ++j) {
            s_sum[base + j] = csum[j];
            s_sq [base + j] = csq[j];
        }
    }
    __syncthreads();

    // ---- normalize + affine, coalesced b128 stores ----
    for (int r = 0; r < 16; ++r) {
        const float mu   = s_sum[r] * (1.0f / 256.0f);
        const float var  = fmaxf(s_sq[r] * (1.0f / 256.0f) - mu * mu, 0.0f);
        const float rstd = rsqrtf(var + 1e-5f);
        const float* yr  = &s_y[r * YS];
        float* orow      = out + (size_t)(rowbase + r) * D_SZ;
#pragma unroll
        for (int hlf = 0; hlf < 2; ++hlf) {
            const int c = hlf * 128 + tid * 4;
            float4 yv = *(const float4*)&yr[c];
            float4 g  = *(const float4*)(gamma + c);
            float4 bt = *(const float4*)(beta + c);
            float4 o;
            o.x = (yv.x - mu) * rstd * g.x + bt.x;
            o.y = (yv.y - mu) * rstd * g.y + bt.y;
            o.z = (yv.z - mu) * rstd * g.z + bt.z;
            o.w = (yv.w - mu) * rstd * g.w + bt.w;
            *(float4*)(orow + c) = o;
        }
    }
}

extern "C" void kernel_launch(void* const* d_in, const int* in_sizes, int n_in,
                              void* d_out, int out_size, void* d_ws, size_t ws_size,
                              hipStream_t stream) {
    // setup_inputs order: x, Wq, bq, Wk, bk, Wv, bv, gamma, beta (all float32).
    // Wq/bq/Wk/bk cancel exactly (softmax over the contracted axis) and are unused.
    (void)in_sizes; (void)n_in; (void)out_size; (void)d_ws; (void)ws_size;
    const float* x     = (const float*)d_in[0];
    const float* Wv    = (const float*)d_in[5];
    const float* bv    = (const float*)d_in[6];
    const float* gamma = (const float*)d_in[7];
    const float* beta  = (const float*)d_in[8];
    float* out = (float*)d_out;

    ufl_attn_ln_kernel<<<dim3(B_SZ / 16), dim3(32), 0, stream>>>(
        x, Wv, bv, gamma, beta, out);
}
